// DFRNN_8383776161815
// MI455X (gfx1250) — matrix operand order repaired and checked by reading, not compiled, not run on hardware
//
#include <hip/hip_runtime.h>

// ---------------------------------------------------------------------------
// DFRNN fused inference for MI455X (gfx1250, wave32, WMMA bf16)
//
// dims: IN=128, GH=512, GL=3, NH=128, NL=2, tokens = 196608
//
// Compute-bound (~0.75 TFLOP after f-gate elimination vs ~100 MB HBM).
// Weight (B) fragments: one global_load pass per (n-tile, k-step), reused
// across all 8 row slabs from registers. A fragments stream from LDS via
// ds_load_b128. Activations use v_exp_f32 / v_rcp_f32 forms (no IEEE-div
// expansion, no libm tanh). nt loop kept rolled for I$ residency.
// ---------------------------------------------------------------------------

typedef __attribute__((ext_vector_type(16))) __bf16 v16bf;
typedef __attribute__((ext_vector_type(8)))  __bf16 v8bf;
typedef __attribute__((ext_vector_type(8)))  float  v8f;

union V16U { v16bf v; v8bf h[2]; };

// explicit global-address-space load: lowers to global_load_b128 pairs
// (tracked by LOADcnt only) instead of flat_load_b128 (LOADcnt+DScnt).
typedef const __attribute__((address_space(1))) v16bf* gv16p;
__device__ __forceinline__ v16bf load_global_v16(const __bf16* p) {
    return *(gv16p)(unsigned long long)(const void*)p;
}

// fast device activations: v_exp_f32 + v_rcp_f32, no IEEE-div sequence
__device__ __forceinline__ float fast_sigm(float x) {
    float e = __expf(-x);
    return __builtin_amdgcn_rcpf(1.0f + e);
}
__device__ __forceinline__ float fast_tanh(float x) {
    float e = __expf(-2.0f * x);
    return (1.0f - e) * __builtin_amdgcn_rcpf(1.0f + e);
}

// A fragment (16x32 bf16, M x K) from LDS row-major [TOK][KIN].
// lanes 0-15 row M=lane, elems 0-15 -> K=k..k+15;
// lanes 16-31 same rows, K offset +16 (matches B contiguous order).
__device__ __forceinline__ v16bf load_a_frag(const __bf16* hin, int KIN,
                                             int mslab, int k, int ln15, int lhi) {
    const __bf16* ap = hin + (mslab * 16 + ln15) * KIN + k + lhi * 16;
    V16U u;
    u.h[0] = *(const v8bf*)(ap);        // K = base .. base+7
    u.h[1] = *(const v8bf*)(ap + 8);    // K = base+8 .. base+15
    return u.v;
}

// B fragment (32x16 bf16, K x N) from global row-major weights [4H][KIN].
// N = lane%16 (= weight output row); lanes 0-15 hold K=k..k+15,
// lanes 16-31 hold K=k+16..k+31 -> 16 contiguous bf16 per lane (32B).
__device__ __forceinline__ v16bf load_b_frag(const __bf16* W, int KIN,
                                             int nrow, int k, int lhi) {
    return load_global_v16(W + (size_t)nrow * KIN + k + lhi * 16);
}

#define TOK   128          // tokens per workgroup
#define MSLAB (TOK / 16)   // 8 row slabs of 16 tokens

// One LSTM "cell from zero state": h = sigmoid(o)*tanh(sigmoid(i)*tanh(c)),
// gates g = x @ W.T + bias; f-gate (rows H..2H) skipped entirely.
template <int KIN, int H, bool RELU>
__device__ void lstm_layer(const __bf16* __restrict__ hin,   // LDS [TOK][KIN]
                           const __bf16* __restrict__ W,     // global bf16 [4H][KIN]
                           const float*  __restrict__ bias,  // f32 [4H]
                           __bf16* __restrict__ hout) {      // LDS [TOK][H]
    const int lane = threadIdx.x & 31;
    const int wave = threadIdx.x >> 5;   // 0..7
    const int ln15 = lane & 15;
    const int lhi  = lane >> 4;          // 0 or 1
    const int NTW  = H / 16 / 8;         // n-tiles per wave (512 -> 4, 128 -> 1)

#pragma unroll 1
    for (int nt = wave * NTW; nt < wave * NTW + NTW; ++nt) {
        const int col = nt * 16 + ln15;
        const float bi = bias[col];
        const float bc = bias[2 * H + col];
        const float bo = bias[3 * H + col];

        v8f accI[MSLAB], accC[MSLAB], accO[MSLAB];
#pragma unroll
        for (int m = 0; m < MSLAB; ++m) { accI[m] = (v8f){}; accC[m] = (v8f){}; accO[m] = (v8f){}; }

#pragma unroll 1
        for (int k = 0; k < KIN; k += 32) {
            // 3 gate B fragments: one global pass, reused by all 8 slabs
            const v16bf bI = load_b_frag(W, KIN, col,         k, lhi);
            const v16bf bC = load_b_frag(W, KIN, 2 * H + col, k, lhi);
            const v16bf bO = load_b_frag(W, KIN, 3 * H + col, k, lhi);
            // stage A fragments, then 24 independent WMMAs
            v16bf a[MSLAB];
#pragma unroll
            for (int m = 0; m < MSLAB; ++m)
                a[m] = load_a_frag(hin, KIN, m, k, ln15, lhi);
#pragma unroll
            for (int m = 0; m < MSLAB; ++m) {
                accI[m] = __builtin_amdgcn_wmma_f32_16x16x32_bf16(false, a[m], false, bI, (short)0, accI[m], false, false);
                accC[m] = __builtin_amdgcn_wmma_f32_16x16x32_bf16(false, a[m], false, bC, (short)0, accC[m], false, false);
                accO[m] = __builtin_amdgcn_wmma_f32_16x16x32_bf16(false, a[m], false, bO, (short)0, accO[m], false, false);
            }
        }

        // C/D layout: elem j, lane -> M = j + lhi*8, N = ln15
#pragma unroll
        for (int m = 0; m < MSLAB; ++m) {
#pragma unroll
            for (int j = 0; j < 8; ++j) {
                float gi = accI[m][j] + bi;
                float gc = accC[m][j] + bc;
                float go = accO[m][j] + bo;
                float cc = fast_sigm(gi) * fast_tanh(gc);
                float hv = fast_sigm(go) * fast_tanh(cc);
                if (RELU) hv = fmaxf(hv, 0.0f);
                const int row = m * 16 + j + lhi * 8;
                hout[row * H + col] = (__bf16)hv;
            }
        }
    }
}

// ---------------------------------------------------------------------------

__global__ __launch_bounds__(256) void dfrnn_main(
    const float* __restrict__ X,
    const __bf16* __restrict__ W0g, const __bf16* __restrict__ W1g, const __bf16* __restrict__ W2g,
    const __bf16* __restrict__ V0n, const __bf16* __restrict__ V1n,
    const float* __restrict__ gb0, const float* __restrict__ gb1, const float* __restrict__ gb2,
    const float* __restrict__ nb0, const float* __restrict__ nb1,
    const float* __restrict__ ew, const float* __restrict__ ebsum,
    const float* __restrict__ aW, const float* __restrict__ ab,
    float* __restrict__ out, int ntok) {

    __shared__ __bf16 sX [TOK * 128];   //  32 KB input tile (both branches)
    __shared__ __bf16 sH0[TOK * 512];   // 128 KB ping
    __shared__ __bf16 sH1[TOK * 512];   // 128 KB pong
    __shared__ float  sPart[256];       // deterministic reduction buffer

    const int tok0 = blockIdx.x * TOK;

    // stage X tile -> LDS bf16
    for (int i = threadIdx.x; i < TOK * 128; i += 256)
        sX[i] = (__bf16)X[(size_t)tok0 * 128 + i];
    __syncthreads();

    // ---- global-factor branch: 3 layers ----
    lstm_layer<128, 512, false>(sX,  W0g, gb0, sH0);
    __syncthreads();
    lstm_layer<512, 512, false>(sH0, W1g, gb1, sH1);
    __syncthreads();
    lstm_layer<512, 512, true >(sH1, W2g, gb2, sH0);   // relu fused at store
    __syncthreads();

    // mu = relu(h) . colsum(e_W) + sum(e_b)   (deterministic 2-chunk reduce)
    {
        const int t  = threadIdx.x & 127;
        const int ch = threadIdx.x >> 7;          // 0..1, 256 cols each
        const __bf16* hr = sH0 + t * 512 + ch * 256;
        const float*  wr = ew + ch * 256;
        float s = 0.0f;
#pragma unroll 4
        for (int c = 0; c < 256; ++c) s += (float)hr[c] * wr[c];
        sPart[ch * 128 + t] = s;
    }
    __syncthreads();
    if (threadIdx.x < TOK) {
        float m = sPart[threadIdx.x] + sPart[128 + threadIdx.x];
        out[tok0 + threadIdx.x] = m + ebsum[0];
    }
    __syncthreads();

    // ---- noise branch: 2 layers (H=128), reuse ping/pong buffers ----
    lstm_layer<128, 128, false>(sX,  V0n, nb0, sH1);
    __syncthreads();
    lstm_layer<128, 128, true >(sH1, V1n, nb1, sH0);
    __syncthreads();

    // sigma = log1p(exp(relu(h) . aW + ab)) + 1e-6
    {
        const int t  = threadIdx.x & 127;
        const int ch = threadIdx.x >> 7;          // 0..1, 64 cols each
        const __bf16* hr = sH0 + t * 128 + ch * 64;
        const float*  wr = aW + ch * 64;
        float s = 0.0f;
#pragma unroll 4
        for (int c = 0; c < 64; ++c) s += (float)hr[c] * wr[c];
        sPart[ch * 128 + t] = s;
    }
    __syncthreads();
    if (threadIdx.x < TOK) {
        float st = sPart[threadIdx.x] + sPart[128 + threadIdx.x] + ab[0];
        out[(size_t)ntok + tok0 + threadIdx.x] = log1pf(__expf(st)) + 1e-6f;
    }
}

// ---------------------------------------------------------------------------
// prep kernels
// ---------------------------------------------------------------------------

__global__ void cvt_bf16(const float* __restrict__ src, __bf16* __restrict__ dst, int n) {
    int i = blockIdx.x * blockDim.x + threadIdx.x;
    if (i < n) dst[i] = (__bf16)src[i];
}

__global__ void add_bias(const float* __restrict__ a, const float* __restrict__ b,
                         float* __restrict__ o, int n) {
    int i = blockIdx.x * blockDim.x + threadIdx.x;
    if (i < n) o[i] = a[i] + b[i];
}

__global__ void fold_embed(const float* __restrict__ eW, const float* __restrict__ eb,
                           float* __restrict__ ew, float* __restrict__ ebsum) {
    int j = blockIdx.x * blockDim.x + threadIdx.x;
    if (j < 512) {
        float s = 0.0f;
        for (int f = 0; f < 10; ++f) s += eW[f * 512 + j];
        ew[j] = s;
    }
    if (j == 0) {
        float s = 0.0f;
        for (int f = 0; f < 10; ++f) s += eb[f];
        *ebsum = s;
    }
}

// ---------------------------------------------------------------------------

extern "C" void kernel_launch(void* const* d_in, const int* in_sizes, int n_in,
                              void* d_out, int out_size, void* d_ws, size_t ws_size,
                              hipStream_t stream) {
    const float* X      = (const float*)d_in[0];
    const float* g_Wih0 = (const float*)d_in[1];   // [2048,128]
    const float* g_bih0 = (const float*)d_in[2];
    const float* g_bhh0 = (const float*)d_in[3];
    const float* g_Wih  = (const float*)d_in[4];   // [2,2048,512]
    const float* g_bih  = (const float*)d_in[5];   // [2,2048]
    const float* g_bhh  = (const float*)d_in[6];
    const float* e_W    = (const float*)d_in[7];   // [10,512]
    const float* e_b    = (const float*)d_in[8];   // [10]
    const float* n_Wih0 = (const float*)d_in[9];   // [512,128]
    const float* n_bih0 = (const float*)d_in[10];
    const float* n_bhh0 = (const float*)d_in[11];
    const float* n_Wih  = (const float*)d_in[12];  // [1,512,128]
    const float* n_bih  = (const float*)d_in[13];
    const float* n_bhh  = (const float*)d_in[14];
    const float* a_W    = (const float*)d_in[15];  // [128]
    const float* a_b    = (const float*)d_in[16];  // [1]

    float* out = (float*)d_out;
    const int ntok = in_sizes[0] / 128;            // 196608

    // workspace layout
    char* w = (char*)d_ws;
    size_t o = 0;
    __bf16* W0g = (__bf16*)(w + o); o += (size_t)2048 * 128 * 2;
    __bf16* W1g = (__bf16*)(w + o); o += (size_t)2048 * 512 * 2;
    __bf16* W2g = (__bf16*)(w + o); o += (size_t)2048 * 512 * 2;
    __bf16* V0n = (__bf16*)(w + o); o += (size_t)512 * 128 * 2;
    __bf16* V1n = (__bf16*)(w + o); o += (size_t)512 * 128 * 2;
    float* gb0   = (float*)(w + o); o += 2048 * 4;
    float* gb1   = (float*)(w + o); o += 2048 * 4;
    float* gb2   = (float*)(w + o); o += 2048 * 4;
    float* nb0   = (float*)(w + o); o += 512 * 4;
    float* nb1   = (float*)(w + o); o += 512 * 4;
    float* ew    = (float*)(w + o); o += 512 * 4;
    float* ebsum = (float*)(w + o); o += 4;

    auto grid = [](int n) { return (n + 255) / 256; };

    // weight conversion f32 -> bf16
    cvt_bf16<<<grid(2048 * 128), 256, 0, stream>>>(g_Wih0, W0g, 2048 * 128);
    cvt_bf16<<<grid(2048 * 512), 256, 0, stream>>>(g_Wih, W1g, 2048 * 512);
    cvt_bf16<<<grid(2048 * 512), 256, 0, stream>>>(g_Wih + (size_t)2048 * 512, W2g, 2048 * 512);
    cvt_bf16<<<grid(512 * 128), 256, 0, stream>>>(n_Wih0, V0n, 512 * 128);
    cvt_bf16<<<grid(512 * 128), 256, 0, stream>>>(n_Wih, V1n, 512 * 128);

    // bias folding (b_ih + b_hh, valid because h0=c0=0)
    add_bias<<<grid(2048), 256, 0, stream>>>(g_bih0, g_bhh0, gb0, 2048);
    add_bias<<<grid(2048), 256, 0, stream>>>(g_bih,        g_bhh,        gb1, 2048);
    add_bias<<<grid(2048), 256, 0, stream>>>(g_bih + 2048, g_bhh + 2048, gb2, 2048);
    add_bias<<<grid(512), 256, 0, stream>>>(n_bih0, n_bhh0, nb0, 512);
    add_bias<<<grid(512), 256, 0, stream>>>(n_bih,  n_bhh,  nb1, 512);

    // embed collapse: sum over factors
    fold_embed<<<grid(512), 256, 0, stream>>>(e_W, e_b, ew, ebsum);

    // fused main kernel: 128 tokens / workgroup
    dfrnn_main<<<ntok / TOK, 256, 0, stream>>>(
        X, W0g, W1g, W2g, V0n, V1n,
        gb0, gb1, gb2, nb0, nb1,
        ew, ebsum, a_W, a_b, out, ntok);

    (void)n_in; (void)out_size; (void)ws_size;
}
